// Graph_Encoder_Norm_32212254720632
// MI455X (gfx1250) — compile-verified, matrix operand order checked by reading, not run on hardware
//
#include <hip/hip_runtime.h>
#include <cstdint>

typedef __attribute__((ext_vector_type(16))) _Float16 v16h;
typedef __attribute__((ext_vector_type(8)))  float    v8f;

#define F_DIM 128
#define C_DIM 64
#define NS_CONV 0.2f
#define NS_ACT  0.01f
#define EPS_N   1e-5f

__device__ __forceinline__ float lrelu(float x, float s) { return x >= 0.f ? x : s * x; }

// order-preserving float <-> uint mapping for atomicMax on signed floats
__device__ __forceinline__ unsigned fflip(float f) {
    unsigned u = __float_as_uint(f);
    return (u & 0x80000000u) ? ~u : (u | 0x80000000u);
}
__device__ __forceinline__ float funflip(unsigned u) {
    u = (u & 0x80000000u) ? (u ^ 0x80000000u) : ~u;
    return __uint_as_float(u);
}

// ---------------- weight fp32 -> fp16 conversion ----------------
__global__ void k_convert_w(const float* __restrict__ inW, const float* __restrict__ linW,
                            _Float16* __restrict__ w16in, _Float16* __restrict__ w16lin) {
    int g = blockIdx.x * blockDim.x + threadIdx.x;
    if (g < C_DIM * F_DIM) w16in[g] = (_Float16)inW[g];
    if (g < 4 * C_DIM * C_DIM) w16lin[g] = (_Float16)linW[g];
}

// ---------------- BN + input Linear(128->64) + LeakyReLU via WMMA ----------------
// 128 threads = 4 waves; block handles 64 node rows; wave w -> output cols [16w,16w+16),
// B fragments held in VGPRs across the 4 M-tiles (16 WMMAs per wave).
__global__ void k_input_gemm(const float* __restrict__ x,
                             const float* __restrict__ bnw, const float* __restrict__ bnb,
                             const float* __restrict__ bnm, const float* __restrict__ bnv,
                             const _Float16* __restrict__ W16, const float* __restrict__ inb,
                             float* __restrict__ h, _Float16* __restrict__ h16, int n) {
    __shared__ _Float16 lA[64 * F_DIM];   // 16 KB
    const int row0 = blockIdx.x * 64;
    const int tid = threadIdx.x;

    if (row0 + 64 <= n) {
        for (int i = tid; i < 64 * F_DIM; i += 128) {
            int r = i >> 7, c = i & 127;
            float v = (x[(row0 + r) * F_DIM + c] - bnm[c]) *
                      rsqrtf(bnv[c] + EPS_N) * bnw[c] + bnb[c];
            lA[i] = (_Float16)v;
        }
    } else {
        for (int i = tid; i < 64 * F_DIM; i += 128) {
            int r = i >> 7, c = i & 127;
            int gr = row0 + r;
            float v = 0.f;
            if (gr < n)
                v = (x[gr * F_DIM + c] - bnm[c]) * rsqrtf(bnv[c] + EPS_N) * bnw[c] + bnb[c];
            lA[i] = (_Float16)v;
        }
    }
    __syncthreads();

    const int wave = tid >> 5;
    const int lane = tid & 31;
    const int hi   = lane >> 4;
    const int row  = lane & 15;
    const int koff = hi ? 8 : 0;
    const int colN = (wave << 4) + (lane & 15);

    // load the wave's B tile (weights) once: 4 K-steps of 16x16x32
    v16h bf[4];
    #pragma unroll
    for (int kk = 0; kk < 4; ++kk) {
        int k0 = kk * 32;
        #pragma unroll
        for (int i = 0; i < 4; ++i) {
            int k1 = k0 + koff + 2 * i;
            int k2 = k0 + 16 + koff + 2 * i;
            bf[kk][2 * i]     = W16[colN * F_DIM + k1];
            bf[kk][2 * i + 1] = W16[colN * F_DIM + k1 + 1];
            bf[kk][8 + 2 * i] = W16[colN * F_DIM + k2];
            bf[kk][9 + 2 * i] = W16[colN * F_DIM + k2 + 1];
        }
    }

    const float bias = inb[colN];
    #pragma unroll
    for (int mt = 0; mt < 4; ++mt) {
        const int rbase = (mt * 16 + row) * F_DIM;
        v8f acc = {};
        #pragma unroll
        for (int kk = 0; kk < 4; ++kk) {
            int k0 = kk * 32;
            v16h a;
            #pragma unroll
            for (int i = 0; i < 4; ++i) {
                int k1 = k0 + koff + 2 * i;
                int k2 = k0 + 16 + koff + 2 * i;
                a[2 * i]     = lA[rbase + k1];
                a[2 * i + 1] = lA[rbase + k1 + 1];
                a[8 + 2 * i] = lA[rbase + k2];
                a[9 + 2 * i] = lA[rbase + k2 + 1];
            }
            acc = __builtin_amdgcn_wmma_f32_16x16x32_f16(false, a, false, bf[kk], (short)0, acc, false, false);
        }
        const int base = row0 + mt * 16 + 8 * hi;
        if (base + 8 <= n) {
            #pragma unroll
            for (int r = 0; r < 8; ++r) {
                float v = lrelu(acc[r] + bias, NS_ACT);
                int idx = (base + r) * C_DIM + colN;   // 32-bit: saddr+scale_offset form
                h[idx]   = v;
                h16[idx] = (_Float16)v;
            }
        } else {
            #pragma unroll
            for (int r = 0; r < 8; ++r) {
                int gr = base + r;
                if (gr < n) {
                    float v = lrelu(acc[r] + bias, NS_ACT);
                    int idx = gr * C_DIM + colN;
                    h[idx]   = v;
                    h16[idx] = (_Float16)v;
                }
            }
        }
    }
}

// ---------------- per-layer z = h @ lin_W^T + lin_b via WMMA (K=64) ----------------
// A tile (64x64 f16, 8 KB) staged with GLOBAL_LOAD_ASYNC_TO_LDS_B128 (ASYNCcnt path).
__global__ void k_layer_gemm(const _Float16* __restrict__ h16,
                             const _Float16* __restrict__ Wl, const float* __restrict__ bl,
                             float* __restrict__ z, int n) {
    __shared__ _Float16 lA[64 * C_DIM];   // 8 KB
    const int row0 = blockIdx.x * 64;
    const int tid = threadIdx.x;

    // async copy: 512 x 16-byte chunks, 4 per thread
    const unsigned ldsBase = (unsigned)(uintptr_t)(&lA[0]);
    const _Float16* gbase = h16 + row0 * C_DIM;
    if (row0 + 64 <= n) {
        #pragma unroll
        for (int j = 0; j < 4; ++j) {
            int chunk = tid + 128 * j;
            unsigned ldsOff = ldsBase + (unsigned)chunk * 16u;
            const void* gaddr = (const void*)(gbase + chunk * 8);
            asm volatile("global_load_async_to_lds_b128 %0, %1, off"
                         :: "v"(ldsOff), "v"(gaddr) : "memory");
        }
    } else {
        #pragma unroll
        for (int j = 0; j < 4; ++j) {
            int chunk = tid + 128 * j;
            int r = chunk >> 3;                  // row within tile
            unsigned ldsOff = ldsBase + (unsigned)chunk * 16u;
            if (row0 + r < n) {
                const void* gaddr = (const void*)(gbase + chunk * 8);
                asm volatile("global_load_async_to_lds_b128 %0, %1, off"
                             :: "v"(ldsOff), "v"(gaddr) : "memory");
            } else {
                float4 zz = {0.f, 0.f, 0.f, 0.f};
                *(float4*)(&lA[chunk * 8]) = zz;
            }
        }
    }
    asm volatile("s_wait_asynccnt 0x0" ::: "memory");
    __syncthreads();

    const int wave = tid >> 5;
    const int lane = tid & 31;
    const int hi   = lane >> 4;
    const int row  = lane & 15;
    const int koff = hi ? 8 : 0;
    const int colN = (wave << 4) + (lane & 15);

    v16h bf[2];
    #pragma unroll
    for (int kk = 0; kk < 2; ++kk) {
        int k0 = kk * 32;
        #pragma unroll
        for (int i = 0; i < 4; ++i) {
            int k1 = k0 + koff + 2 * i;
            int k2 = k0 + 16 + koff + 2 * i;
            bf[kk][2 * i]     = Wl[colN * C_DIM + k1];
            bf[kk][2 * i + 1] = Wl[colN * C_DIM + k1 + 1];
            bf[kk][8 + 2 * i] = Wl[colN * C_DIM + k2];
            bf[kk][9 + 2 * i] = Wl[colN * C_DIM + k2 + 1];
        }
    }

    const float bias = bl[colN];
    #pragma unroll
    for (int mt = 0; mt < 4; ++mt) {
        const int rbase = (mt * 16 + row) * C_DIM;
        v8f acc = {};
        #pragma unroll
        for (int kk = 0; kk < 2; ++kk) {
            int k0 = kk * 32;
            v16h a;
            #pragma unroll
            for (int i = 0; i < 4; ++i) {
                int k1 = k0 + koff + 2 * i;
                int k2 = k0 + 16 + koff + 2 * i;
                a[2 * i]     = lA[rbase + k1];
                a[2 * i + 1] = lA[rbase + k1 + 1];
                a[8 + 2 * i] = lA[rbase + k2];
                a[9 + 2 * i] = lA[rbase + k2 + 1];
            }
            acc = __builtin_amdgcn_wmma_f32_16x16x32_f16(false, a, false, bf[kk], (short)0, acc, false, false);
        }
        const int base = row0 + mt * 16 + 8 * hi;
        if (base + 8 <= n) {
            #pragma unroll
            for (int r = 0; r < 8; ++r)
                z[(base + r) * C_DIM + colN] = acc[r] + bias;
        } else {
            #pragma unroll
            for (int r = 0; r < 8; ++r) {
                int gr = base + r;
                if (gr < n) z[gr * C_DIM + colN] = acc[r] + bias;
            }
        }
    }
}

// ---------------- per-layer scratch init ----------------
__global__ void k_layer_init(float* __restrict__ hnew, const float* __restrict__ convb,
                             unsigned* __restrict__ maxbuf, float* __restrict__ denom,
                             float* __restrict__ stats, int n) {
    int g = blockIdx.x * blockDim.x + threadIdx.x;
    if (g < n * C_DIM) hnew[g] = convb[g & 63];
    if (g < n) { maxbuf[g] = 0u; denom[g] = 0.f; }
    if (g < 2 * C_DIM) stats[g] = 0.f;
}

// ---------------- edge scores: wave per edge ----------------
__global__ void k_edge_score(const float* __restrict__ z, const int* __restrict__ src,
                             const int* __restrict__ dst, const float* __restrict__ attl,
                             float* __restrict__ score, unsigned* __restrict__ maxbuf,
                             int E, int E1) {
    int e = blockIdx.x * (blockDim.x >> 5) + (threadIdx.x >> 5);
    int lane = threadIdx.x & 31;
    if (e >= E1) return;
    if (lane == 0) {
        __builtin_prefetch(src + e + 4096, 0, 0);
        __builtin_prefetch(dst + e + 4096, 0, 0);
    }
    int s = (e < E) ? src[e] : (e - E);
    int d = (e < E) ? dst[e] : (e - E);
    float e0 = lrelu(z[s * C_DIM + lane]      + z[d * C_DIM + lane],      NS_CONV);
    float e1 = lrelu(z[s * C_DIM + 32 + lane] + z[d * C_DIM + 32 + lane], NS_CONV);
    float p = e0 * attl[lane] + e1 * attl[32 + lane];
    #pragma unroll
    for (int m = 16; m >= 1; m >>= 1) p += __shfl_xor(p, m, 32);
    if (lane == 0) {
        score[e] = p;
        atomicMax(&maxbuf[d], fflip(p));
    }
}

// ---------------- exp + denom: thread per edge ----------------
__global__ void k_edge_exp(const int* __restrict__ dst, float* __restrict__ score,
                           const unsigned* __restrict__ maxbuf, float* __restrict__ denom,
                           int E, int E1) {
    int e = blockIdx.x * blockDim.x + threadIdx.x;
    if (e >= E1) return;
    int d = (e < E) ? dst[e] : (e - E);
    float ex = expf(score[e] - funflip(maxbuf[d]));
    score[e] = ex;
    atomicAdd(&denom[d], ex);
}

// ---------------- alpha + scatter aggregate: wave per edge ----------------
__global__ void k_edge_aggr(const float* __restrict__ z, const int* __restrict__ src,
                            const int* __restrict__ dst, const float* __restrict__ score,
                            const float* __restrict__ denom, float* __restrict__ alphas,
                            float* __restrict__ hnew, int E, int E1) {
    int e = blockIdx.x * (blockDim.x >> 5) + (threadIdx.x >> 5);
    int lane = threadIdx.x & 31;
    if (e >= E1) return;
    if (lane == 0) {
        __builtin_prefetch(src + e + 4096, 0, 0);
        __builtin_prefetch(dst + e + 4096, 0, 0);
    }
    int s = (e < E) ? src[e] : (e - E);
    int d = (e < E) ? dst[e] : (e - E);
    float a = score[e] / (denom[d] + 1e-16f);
    if (lane == 0) alphas[e] = a;
    atomicAdd(&hnew[d * C_DIM + lane],      a * z[s * C_DIM + lane]);
    atomicAdd(&hnew[d * C_DIM + 32 + lane], a * z[s * C_DIM + 32 + lane]);
}

// ---------------- GraphNorm stats: per-channel sum + sumsq ----------------
__global__ void k_stats(const float* __restrict__ hnew, float* __restrict__ stats, int n) {
    int g = blockIdx.x * blockDim.x + threadIdx.x;
    int c = g & 63;
    int rstride = (gridDim.x * blockDim.x) >> 6;
    float s = 0.f, s2 = 0.f;
    for (int r = g >> 6; r < n; r += rstride) {
        float v = hnew[r * C_DIM + c];
        s += v; s2 += v * v;
    }
    atomicAdd(&stats[c], s);
    atomicAdd(&stats[64 + c], s2);
}

// ---------------- GraphNorm apply + LeakyReLU ----------------
__global__ void k_norm(const float* __restrict__ hnew, const float* __restrict__ stats,
                       const float* __restrict__ gw, const float* __restrict__ gb,
                       const float* __restrict__ gms, float* __restrict__ h,
                       _Float16* __restrict__ h16, int n) {
    int g = blockIdx.x * blockDim.x + threadIdx.x;
    if (g >= n * C_DIM) return;
    int c = g & 63;
    float invN = 1.f / (float)n;
    float mu  = stats[c] * invN;
    float eh2 = stats[64 + c] * invN;
    float ms  = gms[c];
    float var = eh2 - 2.f * ms * mu * mu + ms * ms * mu * mu;
    float o = hnew[g] - ms * mu;
    float v = gw[c] * o * rsqrtf(var + EPS_N) + gb[c];
    v = lrelu(v, NS_ACT);
    h[g] = v;
    h16[g] = (_Float16)v;
}

// ---------------- write final h twice into d_out ----------------
__global__ void k_writeout(const float* __restrict__ h, float* __restrict__ out, int nc) {
    int g = blockIdx.x * blockDim.x + threadIdx.x;
    if (g >= nc) return;
    float v = h[g];
    out[g] = v;
    out[nc + g] = v;
}

extern "C" void kernel_launch(void* const* d_in, const int* in_sizes, int n_in,
                              void* d_out, int out_size, void* d_ws, size_t ws_size,
                              hipStream_t stream) {
    const float* x    = (const float*)d_in[0];
    const int*   src  = (const int*)d_in[1];
    const int*   dst  = (const int*)d_in[2];
    const float* bnw  = (const float*)d_in[3];
    const float* bnb  = (const float*)d_in[4];
    const float* bnm  = (const float*)d_in[5];
    const float* bnv  = (const float*)d_in[6];
    const float* inW  = (const float*)d_in[7];
    const float* inb  = (const float*)d_in[8];
    const float* linW = (const float*)d_in[9];
    const float* linb = (const float*)d_in[10];
    const float* att  = (const float*)d_in[11];
    const float* cvb  = (const float*)d_in[12];
    const float* gw   = (const float*)d_in[13];
    const float* gb   = (const float*)d_in[14];
    const float* gms  = (const float*)d_in[15];
    float* out = (float*)d_out;

    const int N  = in_sizes[0] / F_DIM;
    const int E  = in_sizes[1];
    const int E1 = E + N;
    const int L  = 4;
    const int NC = N * C_DIM;

    // workspace carve-out
    char* ws = (char*)d_ws;
    size_t off = 0;
    auto carve = [&](size_t bytes) -> void* {
        void* p = ws + off;
        off += (bytes + 255) & ~(size_t)255;
        return p;
    };
    float*    h      = (float*)carve((size_t)NC * 4);
    _Float16* h16    = (_Float16*)carve((size_t)NC * 2);
    float*    z      = (float*)carve((size_t)NC * 4);
    float*    hnew   = (float*)carve((size_t)NC * 4);
    float*    score  = (float*)carve((size_t)E1 * 4);
    unsigned* maxbuf = (unsigned*)carve((size_t)N * 4);
    float*    denom  = (float*)carve((size_t)N * 4);
    float*    stats  = (float*)carve(2 * C_DIM * 4);
    _Float16* w16in  = (_Float16*)carve((size_t)C_DIM * F_DIM * 2);
    _Float16* w16lin = (_Float16*)carve((size_t)L * C_DIM * C_DIM * 2);

    const int TB = 256;
    dim3 blk128(128), blk256(TB);

    k_convert_w<<<(L * C_DIM * C_DIM + TB - 1) / TB, blk256, 0, stream>>>(inW, linW, w16in, w16lin);

    const int gemmBlocks = (N + 63) / 64;
    k_input_gemm<<<gemmBlocks, blk128, 0, stream>>>(x, bnw, bnb, bnm, bnv, w16in, inb, h, h16, N);

    const int edgeBlocks = (E1 + 7) / 8;   // 8 waves per 256-thread block
    const int elemBlocks = (NC + TB - 1) / TB;

    for (int l = 0; l < L; ++l) {
        k_layer_gemm<<<gemmBlocks, blk128, 0, stream>>>(
            h16, w16lin + l * C_DIM * C_DIM, linb + l * C_DIM, z, N);
        k_layer_init<<<elemBlocks, blk256, 0, stream>>>(hnew, cvb + l * C_DIM, maxbuf, denom, stats, N);
        k_edge_score<<<edgeBlocks, blk256, 0, stream>>>(z, src, dst, att + l * C_DIM, score, maxbuf, E, E1);
        k_edge_exp<<<(E1 + TB - 1) / TB, blk256, 0, stream>>>(dst, score, maxbuf, denom, E, E1);
        k_edge_aggr<<<edgeBlocks, blk256, 0, stream>>>(
            z, src, dst, score, denom, out + 2 * (size_t)NC + (size_t)l * E1, hnew, E, E1);
        k_stats<<<256, blk256, 0, stream>>>(hnew, stats, N);
        k_norm<<<elemBlocks, blk256, 0, stream>>>(
            hnew, stats, gw + l * C_DIM, gb + l * C_DIM, gms + l * C_DIM, h, h16, N);
    }

    k_writeout<<<elemBlocks, blk256, 0, stream>>>(h, out, NC);
}
